// MultiHeadAttention_25555055411782
// MI455X (gfx1250) — matrix operand order repaired and checked by reading, not compiled, not run on hardware
//
#include <hip/hip_runtime.h>

#define SEQ    2048
#define BATCHN 4
#define EMB    512
#define NHEAD  8
#define HD     64
#define BHN    (BATCHN * NHEAD)   // 32

typedef __attribute__((ext_vector_type(8)))  __bf16 v8bf;
typedef __attribute__((ext_vector_type(16))) __bf16 v16bf;
typedef __attribute__((ext_vector_type(8)))  float  v8f;
typedef __attribute__((ext_vector_type(4)))  float  v4f;

__device__ __forceinline__ v8f wmma_bf16(v16bf a, v16bf b, v8f c) {
    return __builtin_amdgcn_wmma_f32_16x16x32_bf16(false, a, false, b,
                                                   (short)0, c, false, false);
}

// A-fragment (16x32 bf16): lane holds row lane%16; elements 0..7 = K koff..koff+7,
// elements 8..15 = K koff+16..koff+23, koff = (lane>=16)?8:0. Two 16B chunks.
__device__ __forceinline__ v16bf afrag(const __bf16* __restrict__ p0,
                                       const __bf16* __restrict__ p1) {
    v8bf lo = *(const v8bf*)p0;
    v8bf hi = *(const v8bf*)p1;
    return __builtin_shufflevector(lo, hi, 0, 1, 2, 3, 4, 5, 6, 7,
                                           8, 9, 10, 11, 12, 13, 14, 15);
}

// B-fragment (32x16 bf16): lane holds column lane%16; elements 0..7 = K koff..koff+7,
// elements 8..15 = K koff+16..koff+23, koff = (lane>=16)?8:0. Two 16B chunks.
__device__ __forceinline__ v16bf bfrag(const __bf16* __restrict__ p) {
    v8bf lo = *(const v8bf*)p;
    v8bf hi = *(const v8bf*)(p + 16);
    return __builtin_shufflevector(lo, hi, 0, 1, 2, 3, 4, 5, 6, 7,
                                           8, 9, 10, 11, 12, 13, 14, 15);
}

// =============== Kernel 0: f32 -> bf16 bulk conversion ===============
// Grid is sized to cover the buffer exactly (element counts are multiples of 2048).
__global__ __launch_bounds__(256) void cvt_bf16_kernel(
    const float* __restrict__ src, __bf16* __restrict__ dst) {
    const int i = blockIdx.x * 256 + threadIdx.x;
    const v4f* s = (const v4f*)src + (size_t)i * 2;
    v4f a = s[0], b = s[1];
    v8bf o;
#pragma unroll
    for (int j = 0; j < 4; ++j) {
        o[j]     = (__bf16)a[j];
        o[j + 4] = (__bf16)b[j];
    }
    ((v8bf*)dst)[i] = o;
}

// =============== Kernel 1: fused QKV projection (bf16 WMMA) ===============
__device__ __forceinline__ void store_tile(const v8f& c, int rbase, int obase, int lane,
                                           const float* __restrict__ bias,
                                           __bf16* __restrict__ qws,
                                           __bf16* __restrict__ kws,
                                           __bf16* __restrict__ vtws) {
    const int n    = obase + (lane & 15);     // output-feature column (0..1535)
    const int half = lane >> 4;
    const int h    = n / 192;                 // head
    const int w    = n % 192;
    const int sel  = w / 64;                  // 0=q 1=k 2=v
    const int d    = w % 64;
    const float bv = bias[n];
#pragma unroll
    for (int v = 0; v < 8; ++v) {
        const int row = rbase + v + half * 8; // row in (L*B)
        const int l   = row >> 2;
        const int bi  = row & 3;
        const int bh  = bi * NHEAD + h;
        const float val = c[v] + bv;
        if (sel == 0) {
            qws[((size_t)bh * SEQ + l) * HD + d] = (__bf16)(val * 0.125f);
        } else if (sel == 1) {
            kws[((size_t)bh * SEQ + l) * HD + d] = (__bf16)val;
        } else {
            vtws[((size_t)bh * HD + d) * SEQ + l] = (__bf16)val;
        }
    }
}

__global__ __launch_bounds__(128) void qkv_proj_kernel(
    const __bf16* __restrict__ embb, const __bf16* __restrict__ Wb,
    const float* __restrict__ bias,
    __bf16* __restrict__ qws, __bf16* __restrict__ kws,
    __bf16* __restrict__ vtws) {
    const int lane = threadIdx.x & 31;
    const int wid  = blockIdx.x * 4 + (threadIdx.x >> 5);
    const int mt   = wid % 256;               // 8192/32 row tiles
    const int nt   = wid / 256;               // 1536/32 col tiles
    const int r0   = mt * 32;
    const int o0   = nt * 32;

    const int lrow  = lane & 15;
    const int akoff = (lane >> 4) * 8;
    const int bkoff = (lane >> 4) * 8;

    const __bf16* arow0 = embb + (size_t)(r0 + lrow) * EMB + akoff;
    const __bf16* arow1 = embb + (size_t)(r0 + 16 + lrow) * EMB + akoff;
    const __bf16* brow0 = Wb + (size_t)(o0 + lrow) * EMB + bkoff;
    const __bf16* brow1 = Wb + (size_t)(o0 + 16 + lrow) * EMB + bkoff;

    v8f c00 = {}, c01 = {}, c10 = {}, c11 = {};
    // unroll 4 (not 16): enough loads in flight to cover latency while keeping
    // VGPR allocation < 256 so the compiler avoids s_set_vgpr_msb traffic.
#pragma unroll 4
    for (int k = 0; k < EMB; k += 32) {
        v16bf a0 = afrag(arow0 + k, arow0 + k + 16);
        v16bf a1 = afrag(arow1 + k, arow1 + k + 16);
        v16bf b0 = bfrag(brow0 + k);
        v16bf b1 = bfrag(brow1 + k);
        c00 = wmma_bf16(a0, b0, c00);
        c01 = wmma_bf16(a0, b1, c01);
        c10 = wmma_bf16(a1, b0, c10);
        c11 = wmma_bf16(a1, b1, c11);
    }
    store_tile(c00, r0,      o0,      lane, bias, qws, kws, vtws);
    store_tile(c01, r0,      o0 + 16, lane, bias, qws, kws, vtws);
    store_tile(c10, r0 + 16, o0,      lane, bias, qws, kws, vtws);
    store_tile(c11, r0 + 16, o0 + 16, lane, bias, qws, kws, vtws);
}

// =============== Kernel 2: flash attention (one wave / 16-query tile) ===============
__global__ __launch_bounds__(128) void attn_kernel(
    const __bf16* __restrict__ qws, const __bf16* __restrict__ kws,
    const __bf16* __restrict__ vtws, float* __restrict__ out) {
    __shared__ __bf16 ldsP[4][16 * 32];

    const int lane  = threadIdx.x & 31;
    const int wslot = threadIdx.x >> 5;
    const int wid   = blockIdx.x * 4 + wslot;
    const int bh    = wid >> 7;               // 32 head-batch
    const int qt    = wid & 127;              // 128 query tiles
    const int lrow  = lane & 15;
    const int half  = lane >> 4;
    const int akoff = half * 8;
    const int bkoff = half * 8;

    __bf16* Pl = &ldsP[wslot][0];

    const __bf16* qrow = qws + ((size_t)bh * SEQ + qt * 16 + lrow) * HD + akoff;
    v16bf qf0 = afrag(qrow,      qrow + 16);
    v16bf qf1 = afrag(qrow + 32, qrow + 48);

    v8f o0 = {}, o1 = {}, o2 = {}, o3 = {};
    float mrow[8], lsum[8];
#pragma unroll
    for (int v = 0; v < 8; ++v) { mrow[v] = -3.0e38f; lsum[v] = 0.0f; }

    for (int kb = 0; kb < SEQ; kb += 32) {
        const __bf16* kc0 = kws + ((size_t)bh * SEQ + kb + lrow) * HD + bkoff;
        const __bf16* kc1 = kc0 + 16 * HD;
        v16bf kf00 = bfrag(kc0);
        v16bf kf01 = bfrag(kc0 + 32);
        v16bf kf10 = bfrag(kc1);
        v16bf kf11 = bfrag(kc1 + 32);

        v8f s0 = {}, s1 = {};
        s0 = wmma_bf16(qf0, kf00, s0);
        s0 = wmma_bf16(qf1, kf01, s0);
        s1 = wmma_bf16(qf0, kf10, s1);
        s1 = wmma_bf16(qf1, kf11, s1);

        // online softmax, exact-zero masking (reference: scores==0 -> -inf -> 0)
        float alpha[8];
#pragma unroll
        for (int v = 0; v < 8; ++v) {
            const float a = s0[v], b = s1[v];
            const bool ma = (a == 0.0f), mb = (b == 0.0f);
            float t = fmaxf(ma ? -3.0e38f : a, mb ? -3.0e38f : b);
#pragma unroll
            for (int off = 8; off >= 1; off >>= 1)
                t = fmaxf(t, __shfl_xor(t, off, 32));
            const float mnew = fmaxf(mrow[v], t);
            const float p0 = ma ? 0.0f : __expf(a - mnew);
            const float p1 = mb ? 0.0f : __expf(b - mnew);
            float ps = p0 + p1;
#pragma unroll
            for (int off = 8; off >= 1; off >>= 1)
                ps += __shfl_xor(ps, off, 32);
            const float al = __expf(mrow[v] - mnew);
            alpha[v] = al;
            lsum[v]  = lsum[v] * al + ps;
            mrow[v]  = mnew;
            const int prow = v + half * 8;                 // C-layout row
            Pl[prow * 32 + lrow]      = (__bf16)p0;
            Pl[prow * 32 + 16 + lrow] = (__bf16)p1;
        }
#pragma unroll
        for (int v = 0; v < 8; ++v) {
            o0[v] *= alpha[v]; o1[v] *= alpha[v];
            o2[v] *= alpha[v]; o3[v] *= alpha[v];
        }

        // P A-fragment from LDS (same-wave LDS ops are in order)
        v16bf pf = afrag(Pl + lrow * 32 + akoff, Pl + lrow * 32 + akoff + 16);

        // V B-fragments from transposed V: contiguous along key dim
        const __bf16* vbase = vtws + ((size_t)bh * HD + lrow) * SEQ + kb + bkoff;
        v16bf vf0 = bfrag(vbase);
        v16bf vf1 = bfrag(vbase + 16 * SEQ);
        v16bf vf2 = bfrag(vbase + 32 * SEQ);
        v16bf vf3 = bfrag(vbase + 48 * SEQ);

        o0 = wmma_bf16(pf, vf0, o0);
        o1 = wmma_bf16(pf, vf1, o1);
        o2 = wmma_bf16(pf, vf2, o2);
        o3 = wmma_bf16(pf, vf3, o3);
    }

    const int bi = bh >> 3;
    const int h  = bh & 7;
#pragma unroll
    for (int v = 0; v < 8; ++v) {
        const int l = qt * 16 + v + half * 8;
        const float inv = (lsum[v] > 0.0f) ? (1.0f / lsum[v]) : 0.0f;
        const size_t base = ((size_t)l * BATCHN + bi) * EMB + h * HD + lrow;
        out[base]      = o0[v] * inv;
        out[base + 16] = o1[v] * inv;
        out[base + 32] = o2[v] * inv;
        out[base + 48] = o3[v] * inv;
    }
}

extern "C" void kernel_launch(void* const* d_in, const int* in_sizes, int n_in,
                              void* d_out, int out_size, void* d_ws, size_t ws_size,
                              hipStream_t stream) {
    const float* emb  = (const float*)d_in[0];
    const float* W    = (const float*)d_in[1];
    const float* bias = (const float*)d_in[2];
    float* out = (float*)d_out;

    const size_t qkv_elems = (size_t)BHN * SEQ * HD;     // 4,194,304 each
    const size_t emb_elems = (size_t)SEQ * BATCHN * EMB; // 4,194,304
    const size_t w_elems   = (size_t)3 * EMB * EMB;      // 786,432

    __bf16* qws  = (__bf16*)d_ws;
    __bf16* kws  = qws + qkv_elems;
    __bf16* vtws = kws + qkv_elems;
    __bf16* embb = vtws + qkv_elems;
    __bf16* Wb   = embb + emb_elems;

    cvt_bf16_kernel<<<(int)(emb_elems / 8 / 256), 256, 0, stream>>>(emb, embb);
    cvt_bf16_kernel<<<(int)(w_elems / 8 / 256), 256, 0, stream>>>(W, Wb);
    qkv_proj_kernel<<<3072, 128, 0, stream>>>(embb, Wb, bias, qws, kws, vtws);
    attn_kernel<<<1024, 128, 0, stream>>>(qws, kws, vtws, out);
}
